// PointerGeneratorLM_15908558864806
// MI455X (gfx1250) — matrix operand order repaired and checked by reading, not compiled, not run on hardware
//
#include <hip/hip_runtime.h>
#include <math.h>

// ---------- types ----------
typedef __bf16 bf16_t;
typedef __attribute__((ext_vector_type(16))) __bf16 v16bf;
typedef __attribute__((ext_vector_type(8)))  __bf16 v8bf;
typedef __attribute__((ext_vector_type(8)))  float  v8f;

#define Bdim 2
#define Sdim 1024
#define Ddim 512
#define Vdim 32000
#define BS   (Bdim * Sdim)

// =====================================================================
// bf16 WMMA GEMM:  C[m,n] (+)= sum_k A[m,k] * B[n*sBn + k]  (+ bias[n])
//   A: bf16 row-major (lda), B: K-contiguous rows (stride sBn), C: f32 (ldc)
// Block = 128 thr = 4 waves computes a 64x64 C macro-tile:
//   wave w owns n-tile (blockIdx.x*64 + w*16), all 4 m-tiles (4 accums),
//   so each B fragment feeds 4 back-to-back WMMAs. A 64x32 tile staged in
//   LDS via global_load_async_to_lds_b128 (ASYNCcnt) each K-step.
// Requires: M%64==0, N%64==0, K%32==0 (true for all uses here).
// =====================================================================
__global__ __launch_bounds__(128)
void gemm_bf16_wmma(const bf16_t* __restrict__ A, const bf16_t* __restrict__ Bm,
                    const float* __restrict__ bias, float* __restrict__ C,
                    int M, int N, int K, int lda,
                    long long sBn, int ldc, int accumulate)
{
    __shared__ bf16_t Asmem[64][40];            // 64x32 tile, 80B row stride
                                                // (b128 stores stay 16B aligned)
    const int wave = threadIdx.x >> 5;
    const int lane = threadIdx.x & 31;
    const int m0   = blockIdx.y * 64;
    const int n0   = blockIdx.x * 64 + wave * 16;
    const int m16  = lane & 15;
    const int kh   = lane >> 4;                  // 0 or 1 (lane half)

    const bf16_t* Bbase = Bm + (long long)(n0 + m16) * sBn;

    const int r  = threadIdx.x >> 1;             // 0..63  (A tile row)
    const int cc = (threadIdx.x & 1) * 16;       // 0 or 16 (A tile col chunk)
    const unsigned l0 = (unsigned)(uintptr_t)&Asmem[r][cc];

    v8f acc[4] = {};

    for (int k0 = 0; k0 < K; k0 += 32) {
        // ---- async copy of 64x32 A tile into LDS (2 x b128 per thread) ----
        {
            const unsigned long long gp =
                (unsigned long long)(uintptr_t)(A + (size_t)(m0 + r) * lda + k0 + cc);
            asm volatile("global_load_async_to_lds_b128 %0, %1, off"
                         :: "v"(l0), "v"(gp) : "memory");
            asm volatile("global_load_async_to_lds_b128 %0, %1, off offset:16"
                         :: "v"(l0), "v"(gp) : "memory");
        }

        // ---- B fragment: lane holds N = n0+m16, K = k0+16*kh .. +15 ----
        v16bf bfr;
        {
            const bf16_t* brow = Bbase + k0 + 16 * kh;
            __builtin_prefetch((const void*)(brow + 64), 0, 2);
            v8bf blo = *(const v8bf*)(brow);
            v8bf bhi = *(const v8bf*)(brow + 8);
            #pragma unroll
            for (int j = 0; j < 8; ++j) { bfr[j] = blo[j]; bfr[8 + j] = bhi[j]; }
        }

        asm volatile("s_wait_asynccnt 0x0" ::: "memory");
        __syncthreads();                          // A tile visible to all waves

        // ---- load all 4 A fragments (clause of ds_loads, one wait)... ----
        v16bf af[4];
        #pragma unroll
        for (int mt = 0; mt < 4; ++mt) {
            const bf16_t* arow = &Asmem[mt * 16 + m16][0];
            const int kb = 16 * kh;
            #pragma unroll
            for (int j = 0; j < 8; ++j) {
                af[mt][j]     = arow[kb + j];     // K = kb..kb+7
                af[mt][8 + j] = arow[kb + 8 + j]; // K = kb+8..kb+15
            }
        }
        // ---- ...then 4 back-to-back WMMAs on independent accumulators ----
        #pragma unroll
        for (int mt = 0; mt < 4; ++mt)
            acc[mt] = __builtin_amdgcn_wmma_f32_16x16x32_bf16(
                          false, af[mt], false, bfr, (short)0, acc[mt], false, false);

        __syncthreads();                          // LDS reads done before rewrite
    }

    // ---- store: VGPR i -> M = base + i + 8*kh, N = n0 + m16 ----
    const int col = n0 + m16;
    const float bv = bias ? bias[col] : 0.0f;
    #pragma unroll
    for (int mt = 0; mt < 4; ++mt) {
        #pragma unroll
        for (int i = 0; i < 8; ++i) {
            const size_t off = (size_t)(m0 + mt * 16 + i + 8 * kh) * ldc + col;
            float v = acc[mt][i] + bv;
            if (accumulate) C[off] += v; else C[off] = v;
        }
    }
}

// ---------- elementwise f32 -> bf16 ----------
__global__ void f32_to_bf16(const float* __restrict__ src, bf16_t* __restrict__ dst,
                            size_t n)
{
    size_t i = (size_t)blockIdx.x * blockDim.x + threadIdx.x;
    if (i < n) dst[i] = (bf16_t)src[i];
}

// ---------- embedding gather (f32 + bf16 copies) ----------
__global__ __launch_bounds__(256)
void embed_gather(const int* __restrict__ ids, const float* __restrict__ tok,
                  float* __restrict__ xf, bf16_t* __restrict__ xb)
{
    const int row = blockIdx.x;                  // 0..BS-1
    const int id  = ids[row];
    const float* src = tok + (size_t)id * Ddim;
    for (int k = threadIdx.x; k < Ddim; k += 256) {
        float v = src[k];
        xf[(size_t)row * Ddim + k] = v;
        xb[(size_t)row * Ddim + k] = (bf16_t)v;
    }
}

// ---------- GRU scan: one block per batch, persistent over S ----------
__global__ __launch_bounds__(512)
void gru_scan(const float* __restrict__ gx,      // (B,S,3D) pre-activations (+b_ih)
              const float* __restrict__ W_hh,    // (3D, D)
              const float* __restrict__ b_hh,    // (3D,)
              float* __restrict__ gru_out)       // (B,S,D)
{
    __shared__ float h[Ddim];
    __shared__ float ghs[3 * Ddim];
    const int b = blockIdx.x;
    const int t = threadIdx.x;                   // 0..511
    h[t] = 0.0f;
    __syncthreads();

    for (int s = 0; s < Sdim; ++s) {
        #pragma unroll
        for (int rr = 0; rr < 3; ++rr) {
            const int g = t + rr * Ddim;
            const float* w = W_hh + (size_t)g * Ddim;
            float a = b_hh[g];
            #pragma unroll 8
            for (int k = 0; k < Ddim; ++k) a = fmaf(h[k], w[k], a);
            ghs[g] = a;
        }
        __syncthreads();
        const float* gr = gx + ((size_t)b * Sdim + s) * (3 * Ddim);
        const float xr = gr[t], xz = gr[t + Ddim], xn = gr[t + 2 * Ddim];
        const float hr = ghs[t], hz = ghs[t + Ddim], hn = ghs[t + 2 * Ddim];
        const float rg = 1.0f / (1.0f + __expf(-(xr + hr)));
        const float zg = 1.0f / (1.0f + __expf(-(xz + hz)));
        const float ng = tanhf(xn + rg * hn);
        const float hnew = (1.0f - zg) * ng + zg * h[t];
        __syncthreads();                         // all dot-reads of h done
        h[t] = hnew;
        gru_out[((size_t)b * Sdim + s) * Ddim + t] = hnew;
        __syncthreads();
    }
}

// ---------- attn_in = concat(gru, x_emb): bf16 row-major + per-batch transpose ----------
__global__ __launch_bounds__(256)
void build_attn(const float* __restrict__ gru, const float* __restrict__ xf,
                bf16_t* __restrict__ ab, bf16_t* __restrict__ abT)
{
    const int row = blockIdx.x;                  // b*S + s
    const int b = row >> 10, s = row & (Sdim - 1);
    for (int k = threadIdx.x; k < 2 * Ddim; k += 256) {
        float v = (k < Ddim) ? gru[(size_t)row * Ddim + k]
                             : xf[(size_t)row * Ddim + (k - Ddim)];
        bf16_t bv = (bf16_t)v;
        ab[(size_t)row * (2 * Ddim) + k] = bv;
        abT[((size_t)b * 2 * Ddim + k) * Sdim + s] = bv;   // (2D x S) per batch
    }
}

// ---------- RoPE on Q and K (in-place f32, plus bf16 copies) ----------
__global__ __launch_bounds__(512)
void rope_qk(float* __restrict__ Q, float* __restrict__ Kt,
             bf16_t* __restrict__ Qb, bf16_t* __restrict__ Kb)
{
    const int row = blockIdx.x;                  // 0..BS-1
    const int pos = row & (Sdim - 1);
    const int i   = threadIdx.x;                 // pair index, dim = 2D = 1024
    const float freq  = __expf(-(2.0f * i / 1024.0f) * 9.210340371976184f);
    float sn, cs;
    __sincosf((float)pos * freq, &sn, &cs);
    const size_t o = (size_t)row * 1024 + 2 * i;
    float q0 = Q[o], q1 = Q[o + 1];
    float k0 = Kt[o], k1 = Kt[o + 1];
    float nq0 = q0 * cs - q1 * sn, nq1 = q1 * cs + q0 * sn;
    float nk0 = k0 * cs - k1 * sn, nk1 = k1 * cs + k0 * sn;
    Q[o] = nq0;  Q[o + 1] = nq1;
    Kt[o] = nk0; Kt[o + 1] = nk1;
    Qb[o] = (bf16_t)nq0; Qb[o + 1] = (bf16_t)nq1;
    Kb[o] = (bf16_t)nk0; Kb[o + 1] = (bf16_t)nk1;
}

// ---------- causal softmax over rows of scores (in-place) + bf16 probs ----------
__global__ __launch_bounds__(256)
void causal_softmax(float* __restrict__ scores, bf16_t* __restrict__ pb)
{
    __shared__ float red[256];
    const int row = blockIdx.x;                  // b*S + i
    const int i   = row & (Sdim - 1);
    const int t   = threadIdx.x;
    float* sr  = scores + (size_t)row * Sdim;
    bf16_t* pr = pb + (size_t)row * Sdim;
    const float scale = 0.03125f;                // 1/sqrt(1024)

    float mx = -3.4e38f;
    for (int j = t; j < Sdim; j += 256)
        if (j <= i) mx = fmaxf(mx, sr[j] * scale);
    red[t] = mx; __syncthreads();
    for (int s2 = 128; s2 > 0; s2 >>= 1) {
        if (t < s2) red[t] = fmaxf(red[t], red[t + s2]);
        __syncthreads();
    }
    mx = red[0]; __syncthreads();

    float sum = 0.0f;
    for (int j = t; j < Sdim; j += 256) {
        float e = (j <= i) ? __expf(sr[j] * scale - mx) : 0.0f;
        sr[j] = e;
        sum += e;
    }
    red[t] = sum; __syncthreads();
    for (int s2 = 128; s2 > 0; s2 >>= 1) {
        if (t < s2) red[t] += red[t + s2];
        __syncthreads();
    }
    const float inv = 1.0f / red[0];

    for (int j = t; j < Sdim; j += 256) {
        float p = sr[j] * inv;
        sr[j] = p;
        pr[j] = (bf16_t)p;
    }
}

// ---------- copy gru_out into combined[:, :D] ----------
__global__ __launch_bounds__(256)
void gru_to_combined(const float* __restrict__ gru, float* __restrict__ comb)
{
    const int row = blockIdx.x;
    for (int k = threadIdx.x; k < Ddim; k += 256)
        comb[(size_t)row * (3 * Ddim) + k] = gru[(size_t)row * Ddim + k];
}

// ---------- gate = sigmoid(combined @ Wgate^T + bgate) ----------
__global__ __launch_bounds__(256)
void gate_kernel(const float* __restrict__ comb, const float* __restrict__ Wg,
                 const float* __restrict__ bg, float* __restrict__ gate)
{
    __shared__ float red[256];
    const int row = blockIdx.x;
    const int t   = threadIdx.x;
    const float* cr = comb + (size_t)row * (3 * Ddim);
    float a = 0.0f;
    for (int k = t; k < 3 * Ddim; k += 256) a = fmaf(cr[k], Wg[k], a);
    red[t] = a; __syncthreads();
    for (int s2 = 128; s2 > 0; s2 >>= 1) {
        if (t < s2) red[t] += red[t + s2];
        __syncthreads();
    }
    if (t == 0) gate[row] = 1.0f / (1.0f + __expf(-(red[0] + bg[0])));
}

// ---------- softmax over vocab (in d_out) scaled by gate ----------
__global__ __launch_bounds__(256)
void vocab_finalize(float* __restrict__ out, const float* __restrict__ gate)
{
    __shared__ float red[256];
    const int row = blockIdx.x;
    const int t   = threadIdx.x;
    float* lr = out + (size_t)row * Vdim;

    float mx = -3.4e38f;
    for (int j = t; j < Vdim; j += 256) mx = fmaxf(mx, lr[j]);
    red[t] = mx; __syncthreads();
    for (int s2 = 128; s2 > 0; s2 >>= 1) {
        if (t < s2) red[t] = fmaxf(red[t], red[t + s2]);
        __syncthreads();
    }
    mx = red[0]; __syncthreads();

    float sum = 0.0f;
    for (int j = t; j < Vdim; j += 256) sum += __expf(lr[j] - mx);
    red[t] = sum; __syncthreads();
    for (int s2 = 128; s2 > 0; s2 >>= 1) {
        if (t < s2) red[t] += red[t + s2];
        __syncthreads();
    }
    const float g = gate[row];
    const float inv = g / red[0];
    for (int j = t; j < Vdim; j += 256) lr[j] = __expf(lr[j] - mx) * inv;
}

// ---------- scatter copy distribution: out[b,i,x[b,j]] += (1-g)*P[b,i,j] ----------
__global__ __launch_bounds__(256)
void copy_scatter(const float* __restrict__ probs, const int* __restrict__ ids,
                  const float* __restrict__ gate, float* __restrict__ out)
{
    const int row = blockIdx.x;                  // b*S + i
    const int b   = row >> 10;
    const int i   = row & (Sdim - 1);
    const float g1 = 1.0f - gate[row];
    const float* pr = probs + (size_t)row * Sdim;
    const int* xb = ids + (size_t)b * Sdim;
    float* orow = out + (size_t)row * Vdim;
    for (int j = threadIdx.x; j <= i; j += 256)
        atomicAdd(orow + xb[j], g1 * pr[j]);
}

// =====================================================================
extern "C" void kernel_launch(void* const* d_in, const int* in_sizes, int n_in,
                              void* d_out, int out_size, void* d_ws, size_t ws_size,
                              hipStream_t stream)
{
    (void)in_sizes; (void)n_in; (void)out_size; (void)ws_size;
    const int*   x     = (const int*)  d_in[0];
    const float* tok   = (const float*)d_in[1];
    const float* pose  = (const float*)d_in[2];
    const float* W_ih  = (const float*)d_in[3];
    const float* W_hh  = (const float*)d_in[4];
    const float* b_ih  = (const float*)d_in[5];
    const float* b_hh  = (const float*)d_in[6];
    const float* Wq    = (const float*)d_in[7];
    const float* Wk    = (const float*)d_in[8];
    const float* Wqp   = (const float*)d_in[9];
    const float* Wgen  = (const float*)d_in[10];
    const float* bgen  = (const float*)d_in[11];
    const float* Wgate = (const float*)d_in[12];
    const float* bgate = (const float*)d_in[13];
    float* out = (float*)d_out;

    // ---- workspace carve-up (256B aligned) ----
    char* wsp = (char*)d_ws;
    size_t off = 0;
    auto alloc = [&](size_t bytes) -> void* {
        void* p = wsp + off;
        off += (bytes + 255) & ~(size_t)255;
        return p;
    };
    float*  xemb    = (float*) alloc((size_t)BS * Ddim * 4);
    bf16_t* xemb_bf = (bf16_t*)alloc((size_t)BS * Ddim * 2);
    float*  gx      = (float*) alloc((size_t)BS * 3 * Ddim * 4);
    float*  gru     = (float*) alloc((size_t)BS * Ddim * 4);
    bf16_t* attn_bf = (bf16_t*)alloc((size_t)BS * 2 * Ddim * 2);
    bf16_t* attn_T  = (bf16_t*)alloc((size_t)Bdim * 2 * Ddim * Sdim * 2);
    float*  Qf      = (float*) alloc((size_t)BS * 2 * Ddim * 4);
    float*  Kf      = (float*) alloc((size_t)BS * 2 * Ddim * 4);
    float*  Qp      = (float*) alloc((size_t)BS * Ddim * 4);
    bf16_t* Qb      = (bf16_t*)alloc((size_t)BS * 2 * Ddim * 2);
    bf16_t* Kb      = (bf16_t*)alloc((size_t)BS * 2 * Ddim * 2);
    bf16_t* Qpb     = (bf16_t*)alloc((size_t)BS * Ddim * 2);
    float*  sc      = (float*) alloc((size_t)Bdim * Sdim * Sdim * 4);
    bf16_t* pb      = (bf16_t*)alloc((size_t)Bdim * Sdim * Sdim * 2);
    float*  comb    = (float*) alloc((size_t)BS * 3 * Ddim * 4);
    bf16_t* comb_bf = (bf16_t*)alloc((size_t)BS * 3 * Ddim * 2);
    float*  gate    = (float*) alloc((size_t)BS * 4);
    bf16_t* pos_bf  = (bf16_t*)alloc((size_t)Sdim * Ddim * 2);
    bf16_t* Wih_bf  = (bf16_t*)alloc((size_t)3 * Ddim * Ddim * 2);
    bf16_t* Wq_bf   = (bf16_t*)alloc((size_t)2 * Ddim * 2 * Ddim * 2);
    bf16_t* Wk_bf   = (bf16_t*)alloc((size_t)2 * Ddim * 2 * Ddim * 2);
    bf16_t* Wqp_bf  = (bf16_t*)alloc((size_t)Ddim * 2 * Ddim * 2);
    bf16_t* Wgen_bf = (bf16_t*)alloc((size_t)Vdim * 3 * Ddim * 2);

    auto cvt = [&](const float* s, bf16_t* d, size_t n) {
        f32_to_bf16<<<dim3((unsigned)((n + 255) / 256)), dim3(256), 0, stream>>>(s, d, n);
    };
    auto gemm = [&](const bf16_t* A, const bf16_t* Bm, const float* bias, float* C,
                    int M, int N, int K, int lda, long long sBn, int ldc, int acc) {
        gemm_bf16_wmma<<<dim3(N / 64, M / 64), dim3(128), 0, stream>>>(
            A, Bm, bias, C, M, N, K, lda, sBn, ldc, acc);
    };

    // 1) weight / pos-emb conversions to bf16
    cvt(W_ih, Wih_bf, (size_t)3 * Ddim * Ddim);
    cvt(Wq,   Wq_bf,  (size_t)2 * Ddim * 2 * Ddim);
    cvt(Wk,   Wk_bf,  (size_t)2 * Ddim * 2 * Ddim);
    cvt(Wqp,  Wqp_bf, (size_t)Ddim * 2 * Ddim);
    cvt(Wgen, Wgen_bf,(size_t)Vdim * 3 * Ddim);
    cvt(pose, pos_bf, (size_t)Sdim * Ddim);          // first S rows

    // 2) embedding gather
    embed_gather<<<dim3(BS), dim3(256), 0, stream>>>(x, tok, xemb, xemb_bf);

    // 3) gx = x_emb @ W_ih^T + b_ih   (BS x 1536)
    gemm(xemb_bf, Wih_bf, b_ih, gx, BS, 3 * Ddim, Ddim, Ddim, Ddim, 3 * Ddim, 0);

    // 4) GRU scan (one persistent block per batch)
    gru_scan<<<dim3(Bdim), dim3(512), 0, stream>>>(gx, W_hh, b_hh, gru);

    // 5) attn_in = concat(gru, x_emb): bf16 row-major + (2D x S) transpose
    build_attn<<<dim3(BS), dim3(256), 0, stream>>>(gru, xemb, attn_bf, attn_T);

    // 6) Q, K, Qpos projections
    gemm(attn_bf, Wq_bf,  nullptr, Qf, BS, 2 * Ddim, 2 * Ddim, 2 * Ddim, 2 * Ddim, 2 * Ddim, 0);
    gemm(attn_bf, Wk_bf,  nullptr, Kf, BS, 2 * Ddim, 2 * Ddim, 2 * Ddim, 2 * Ddim, 2 * Ddim, 0);
    gemm(attn_bf, Wqp_bf, nullptr, Qp, BS, Ddim,     2 * Ddim, 2 * Ddim, 2 * Ddim, Ddim, 0);

    // 7) RoPE on Q,K (+ bf16); bf16 copy of Qpos
    rope_qk<<<dim3(BS), dim3(512), 0, stream>>>(Qf, Kf, Qb, Kb);
    cvt(Qp, Qpb, (size_t)BS * Ddim);

    // 8) scores = Q@K^T + Qpos@pos_emb^T  (per batch)
    for (int b = 0; b < Bdim; ++b) {
        const size_t ro = (size_t)b * Sdim;
        gemm(Qb + ro * 2 * Ddim, Kb + ro * 2 * Ddim, nullptr, sc + ro * Sdim,
             Sdim, Sdim, 2 * Ddim, 2 * Ddim, 2 * Ddim, Sdim, 0);
        gemm(Qpb + ro * Ddim, pos_bf, nullptr, sc + ro * Sdim,
             Sdim, Sdim, Ddim, Ddim, Ddim, Sdim, 1);
    }

    // 9) causal softmax (scale 1/sqrt(2D) folded in)
    causal_softmax<<<dim3(BS), dim3(256), 0, stream>>>(sc, pb);

    // 10) combined[:, :D] = gru ; combined[:, D:] = probs @ attn_in (via attn_T)
    gru_to_combined<<<dim3(BS), dim3(256), 0, stream>>>(gru, comb);
    for (int b = 0; b < Bdim; ++b) {
        const size_t ro = (size_t)b * Sdim;
        gemm(pb + ro * Sdim, attn_T + (size_t)b * 2 * Ddim * Sdim, nullptr,
             comb + ro * 3 * Ddim + Ddim,
             Sdim, 2 * Ddim, Sdim, Sdim, Sdim, 3 * Ddim, 0);
    }
    cvt(comb, comb_bf, (size_t)BS * 3 * Ddim);

    // 11) gate = sigmoid(combined @ Wgate^T + bgate)
    gate_kernel<<<dim3(BS), dim3(256), 0, stream>>>(comb, Wgate, bgate, gate);

    // 12) logits = combined @ Wgen^T + bgen  -> d_out (used as scratch)
    gemm(comb_bf, Wgen_bf, bgen, out, BS, Vdim, 3 * Ddim, 3 * Ddim, 3 * Ddim, Vdim, 0);

    // 13) out = gate * softmax(logits)
    vocab_finalize<<<dim3(BS), dim3(256), 0, stream>>>(out, gate);

    // 14) out += (1-gate) * copy distribution (atomic scatter)
    copy_scatter<<<dim3(BS), dim3(256), 0, stream>>>(sc, x, gate, out);
}
